// RhoNN_31439160607385
// MI455X (gfx1250) — compile-verified, matrix operand order checked by reading, not run on hardware
//
#include <hip/hip_runtime.h>
#include <hip/hip_bf16.h>

// ---------------- problem constants (from reference) ----------------
#define NA   4096            // N_AGENT
#define NN   8192            // total nodes
#define EP   43690           // edges per type
#define ET   131070          // total edges (3 * EP)
#define FD   1024            // feature dim
#define HD   1024            // hidden dim

typedef __attribute__((ext_vector_type(16))) __bf16 v16bf;
typedef __attribute__((ext_vector_type(8)))  float  v8f;
typedef __attribute__((ext_vector_type(4)))  unsigned v4u;
typedef __attribute__((ext_vector_type(8)))  int    v8i;
typedef __attribute__((ext_vector_type(4)))  int    v4i;

union ABfrag { v16bf v; v4u q[2]; };

__device__ __forceinline__ unsigned short f2bf(float f) {
    unsigned u = __float_as_uint(f);
    unsigned r = u + 0x7FFFu + ((u >> 16) & 1u);   // round-to-nearest-even
    return (unsigned short)(r >> 16);
}
__device__ __forceinline__ unsigned pack2bf(float lo, float hi) {
    const unsigned ulo = __float_as_uint(lo) + 0x8000u;   // round-half-up
    const unsigned uhi = __float_as_uint(hi) + 0x8000u;
    return __builtin_amdgcn_perm(uhi, ulo, 0x07060302u);  // {uhi.hi16, ulo.hi16}
}
__device__ __forceinline__ float bf2f(unsigned short h) {
    return __uint_as_float(((unsigned)h) << 16);
}

// ---------------------------------------------------------------------
// TDM: DMA one [128 rows x 32 k] bf16 tile (row-contiguous K) into LDS.
// D# per CDNA5 ISA section 8.3/8.4. g1 (dims/tile) is loop-invariant and
// built once per operand; only g0 (lds / global address) varies per issue.
// ---------------------------------------------------------------------
__device__ __forceinline__ v8i tdm_make_g1(unsigned remRows) {
    return v8i{ (int)(1u << 16),                         // data_size = 2 bytes
                (int)(1024u << 16),                      // tensor_dim0 = 1024 (lo16)
                (int)((remRows & 0xFFFFu) << 16),        // dim0 hi=0 | tensor_dim1 lo16
                (int)((remRows >> 16) | (32u << 16)),    // tensor_dim1 hi | tile_dim0=32
                128,                                     // tile_dim1 = 128 (tile_dim2 = 0)
                1024,                                    // tensor_dim0_stride (elements)
                0, 0 };
}
__device__ __forceinline__ void tdm_issue(unsigned ldsOff,
                                          const unsigned short* gptr, v8i g1) {
    const unsigned long long ga = (unsigned long long)(size_t)gptr;
    const v4u g0 = { 1u,                                 // count=1, user desc
                     ldsOff,                             // lds_addr [63:32]
                     (unsigned)ga,                       // global_addr lo
                     ((unsigned)(ga >> 32) & 0x01FFFFFFu) | 0x80000000u }; // hi | type=2
    const v4i gz4 = {0, 0, 0, 0};
    const v8i gz8 = {0, 0, 0, 0, 0, 0, 0, 0};
    __builtin_amdgcn_tensor_load_to_lds(g0, g1, gz4, gz4, gz8, 0);
}

// =====================================================================
// bf16 WMMA GEMM:  C[M,1024] = A[M,1024](bf16) @ Bt[1024,1024](bf16, n-major) + bias
// Workgroup 256 = 8 waves (4x2); 128x128 block tile; wave tile 32x64; K step 32.
// Double-buffered LDS fed by TDM tensor_load_to_lds (wave 0 issues).
// =====================================================================
template <bool BF16OUT>
__global__ __launch_bounds__(256) void hgt_gemm(const unsigned short* __restrict__ A,
                                                const unsigned short* __restrict__ Bt,
                                                const float* __restrict__ bias,
                                                void* __restrict__ Cout, int M) {
    __shared__ __align__(16) unsigned lA[2][128 * 16];  // 128 rows x 32 k (bf16 pairs)
    __shared__ __align__(16) unsigned lB[2][128 * 16];  // 128 n    x 32 k (bf16 pairs)

    const int tid  = threadIdx.x;
    const int lane = tid & 31;
    const int wave = tid >> 5;
    const int wm   = wave >> 1;         // 0..3  (M direction)
    const int wn   = wave & 1;          // 0..1  (N direction)
    const int half = lane >> 4;
    const int r    = lane & 15;

    const int rowBase = blockIdx.y * 128;
    const int colBase = blockIdx.x * 128;

    const unsigned ldsA[2] = { (unsigned)(size_t)(void*)&lA[0][0],
                               (unsigned)(size_t)(void*)&lA[1][0] };
    const unsigned ldsB[2] = { (unsigned)(size_t)(void*)&lB[0][0],
                               (unsigned)(size_t)(void*)&lB[1][0] };

    const v8i g1A = tdm_make_g1((unsigned)(M - rowBase));
    const v8i g1B = tdm_make_g1((unsigned)(HD - colBase));
    const unsigned short* aPtr = A + (size_t)rowBase * FD;
    const unsigned short* bPtr = Bt + (size_t)colBase * FD;

    auto issue = [&](int buf, int kb) {
        tdm_issue(ldsA[buf], aPtr + kb, g1A);
        tdm_issue(ldsB[buf], bPtr + kb, g1B);
    };

    v8f acc[2][4];
    const v8f vzero = {0.f,0.f,0.f,0.f,0.f,0.f,0.f,0.f};
#pragma unroll
    for (int i = 0; i < 2; ++i)
#pragma unroll
        for (int j = 0; j < 4; ++j) acc[i][j] = vzero;

    if (wave == 0) issue(0, 0);
    __builtin_amdgcn_s_wait_tensorcnt(0);
    __syncthreads();

    const int NIT = FD / 32;
#pragma unroll 2
    for (int it = 0; it < NIT; ++it) {
        const int cur = it & 1;
        if (wave == 0 && it + 1 < NIT) issue(1 - cur, (it + 1) * 32);

        const unsigned* la = lA[cur];
        const unsigned* lb = lB[cur];
        ABfrag a[2], b[4];
#pragma unroll
        for (int tm = 0; tm < 2; ++tm) {
            const v4u* p = (const v4u*)(la + (wm * 32 + tm * 16 + r) * 16 + half * 4);
            a[tm].q[0] = p[0];      // k-pairs half*4 .. half*4+3
            a[tm].q[1] = p[2];      // k-pairs 8+half*4 ..
        }
#pragma unroll
        for (int tn = 0; tn < 4; ++tn) {
            const v4u* p = (const v4u*)(lb + (wn * 64 + tn * 16 + r) * 16 + half * 4);
            b[tn].q[0] = p[0];
            b[tn].q[1] = p[2];
        }
#pragma unroll
        for (int tm = 0; tm < 2; ++tm)
#pragma unroll
            for (int tn = 0; tn < 4; ++tn)
                acc[tm][tn] = __builtin_amdgcn_wmma_f32_16x16x32_bf16(
                    false, a[tm].v, false, b[tn].v, (short)0, acc[tm][tn],
                    false, false);

        __builtin_amdgcn_s_wait_tensorcnt(0);   // no-op for waves 1..7
        __syncthreads();
    }

    // ---- epilogue: bias add + store (C layout: VGPR v -> M = v + half*8, N = r) ----
#pragma unroll
    for (int tm = 0; tm < 2; ++tm) {
#pragma unroll
        for (int tn = 0; tn < 4; ++tn) {
            const int n  = colBase + wn * 64 + tn * 16 + r;
            const float bv = bias[n];
#pragma unroll
            for (int v = 0; v < 8; ++v) {
                const int mrow = rowBase + wm * 32 + tm * 16 + half * 8 + v;
                if (mrow < M) {
                    const float val = acc[tm][tn][v] + bv;
                    if (BF16OUT)
                        ((unsigned short*)Cout)[(size_t)mrow * HD + n] = f2bf(val);
                    else
                        ((float*)Cout)[(size_t)mrow * HD + n] = val;
                }
            }
        }
    }
}

// =====================================================================
// conversion / transpose pre-passes
// =====================================================================
// fp32 -> bf16, n must be a multiple of 4
__global__ __launch_bounds__(256) void hgt_cvt(const float* __restrict__ src,
                                               unsigned short* __restrict__ dst, long n) {
    const long i = ((long)blockIdx.x * 256 + threadIdx.x) * 4;
    if (i >= n) return;
    const float4 f = *(const float4*)(src + i);
    uint2 o;
    o.x = pack2bf(f.x, f.y);
    o.y = pack2bf(f.z, f.w);
    *(uint2*)(dst + i) = o;
}

// W [1024 k x 1024 n] fp32 row-major  ->  WT [1024 n x 1024 k] bf16 (batched)
__global__ __launch_bounds__(256) void hgt_cvt_transpose(const float* __restrict__ W,
                                                         unsigned short* __restrict__ WT) {
    __shared__ unsigned short t[32][33];
    const float* src = W + (size_t)blockIdx.z * FD * HD;
    unsigned short* dst = WT + (size_t)blockIdx.z * FD * HD;
    const int n0 = blockIdx.x * 32, k0 = blockIdx.y * 32;
    const int tx = threadIdx.x & 31, ty = threadIdx.x >> 5;   // 32 x 8
#pragma unroll
    for (int i = 0; i < 32; i += 8)
        t[ty + i][tx] = f2bf(src[(size_t)(k0 + ty + i) * HD + n0 + tx]);
    __syncthreads();
#pragma unroll
    for (int i = 0; i < 32; i += 8)
        dst[(size_t)(n0 + ty + i) * FD + k0 + tx] = t[tx][ty + i];
}

// =====================================================================
// Attention edge phase
// =====================================================================
__global__ void hgt_init(float* __restrict__ m, float* __restrict__ denom, int n) {
    int i = blockIdx.x * blockDim.x + threadIdx.x;
    if (i < n) { m[i] = -3.402823466e38f; denom[i] = 0.f; }
}

// one wave per edge: score = dot(w3f[recv], w4f[send] + w5f[e]) / 8 ; atomic max
__global__ __launch_bounds__(256) void hgt_score(const float* __restrict__ w3f,
                                                 const float* __restrict__ w4f,
                                                 const unsigned short* __restrict__ w5f,
                                                 const int* __restrict__ recv,
                                                 const int* __restrict__ send,
                                                 float* __restrict__ score,
                                                 float* __restrict__ m, int E) {
    const int e    = blockIdx.x * 8 + (threadIdx.x >> 5);
    const int lane = threadIdx.x & 31;
    if (e >= E) return;
    const int ri = recv[e];
    const float4* q  = (const float4*)(w3f + (size_t)ri * HD);
    const float4* kv = (const float4*)(w4f + (size_t)send[e] * HD);
    const uint2*  w5 = (const uint2*)(w5f + (size_t)e * HD);
    float s = 0.f;
#pragma unroll 2
    for (int i = lane; i < HD / 4; i += 32) {
        const float4 qv = q[i];
        const float4 kk = kv[i];
        const uint2  wv = w5[i];
        s += qv.x * (kk.x + __uint_as_float(wv.x << 16));
        s += qv.y * (kk.y + __uint_as_float(wv.x & 0xFFFF0000u));
        s += qv.z * (kk.z + __uint_as_float(wv.y << 16));
        s += qv.w * (kk.w + __uint_as_float(wv.y & 0xFFFF0000u));
    }
#pragma unroll
    for (int off = 16; off > 0; off >>= 1) s += __shfl_xor(s, off, 32);
    if (lane == 0) {
        s *= 0.125f;                               // 1/sqrt(64)
        score[e] = s;
        if (s >= 0.f) atomicMax((int*)&m[ri], __float_as_int(s));
        else          atomicMin((unsigned*)&m[ri], (unsigned)__float_as_int(s));
    }
}

// ex = exp(score - m[recv]); denom[recv] += ex   (score overwritten in place)
__global__ void hgt_exp(float* __restrict__ score, const float* __restrict__ m,
                        float* __restrict__ denom, const int* __restrict__ recv, int E) {
    int e = blockIdx.x * blockDim.x + threadIdx.x;
    if (e >= E) return;
    const int ri = recv[e];
    const float ex = __expf(score[e] - m[ri]);
    score[e] = ex;
    atomicAdd(&denom[ri], ex);
}

// one wave per edge: agg[recv] += alpha * (w2f[send] + w5f[e])
__global__ __launch_bounds__(256) void hgt_msg(const float* __restrict__ w2f,
                                               const unsigned short* __restrict__ w5f,
                                               const float* __restrict__ ex,
                                               const float* __restrict__ denom,
                                               const int* __restrict__ recv,
                                               const int* __restrict__ send,
                                               float* __restrict__ agg, int E) {
    const int e    = blockIdx.x * 8 + (threadIdx.x >> 5);
    const int lane = threadIdx.x & 31;
    if (e >= E) return;
    const int ri = recv[e], si = send[e];
    const float d     = denom[ri];
    const float alpha = ex[e] / (d == 0.f ? 1.f : d);
    const float* w2          = w2f + (size_t)si * HD;
    const unsigned short* w5 = w5f + (size_t)e * HD;
    float* dst               = agg + (size_t)ri * HD;
#pragma unroll 4
    for (int h = lane; h < HD; h += 32)
        atomicAdd(&dst[h], alpha * (w2[h] + bf2f(w5[h])));
}

// =====================================================================
// host-side layer driver
// =====================================================================
static void run_layer(const unsigned short* Xbf /* [NN x FD] bf16 */,
                      const unsigned short* EFbf /* [ET x FD] bf16 */,
                      const float* Wn, const float* bn,
                      const float* We, const float* be,
                      const int* recv, const int* send,
                      unsigned short* WnT, unsigned short* WeT,
                      float* agg /*= w1f table, becomes layer output*/,
                      float* w2f, float* w3f, float* w4f,
                      unsigned short* w5f, float* score, float* m, float* denom,
                      hipStream_t stream) {
    const dim3 blk(256);
    // convert + transpose this layer's weights to bf16 n-major
    hgt_cvt_transpose<<<dim3(32, 32, 8), blk, 0, stream>>>(Wn, WnT);
    hgt_cvt_transpose<<<dim3(32, 32, 3), blk, 0, stream>>>(We, WeT);

    float* tables[4] = {agg, w2f, w3f, w4f};
    const dim3 gN(HD / 128, NA / 128);               // 8 x 32
    for (int t = 0; t < 2; ++t) {
        const unsigned short* X = Xbf + (size_t)t * NA * FD;
        for (int w = 0; w < 4; ++w) {
            hgt_gemm<false><<<gN, blk, 0, stream>>>(
                X, WnT + (size_t)(t * 4 + w) * FD * HD, bn + (size_t)(t * 4 + w) * HD,
                tables[w] + (size_t)t * NA * HD, NA);
        }
    }
    const dim3 gE(HD / 128, (EP + 127) / 128);        // 8 x 342
    for (int t = 0; t < 3; ++t) {
        hgt_gemm<true><<<gE, blk, 0, stream>>>(
            EFbf + (size_t)t * EP * FD, WeT + (size_t)t * FD * HD, be + (size_t)t * HD,
            w5f + (size_t)t * EP * HD, EP);
    }
    hgt_init<<<dim3((NN + 255) / 256), blk, 0, stream>>>(m, denom, NN);
    hgt_score<<<dim3((ET + 7) / 8), blk, 0, stream>>>(w3f, w4f, w5f, recv, send,
                                                      score, m, ET);
    hgt_exp<<<dim3((ET + 255) / 256), blk, 0, stream>>>(score, m, denom, recv, ET);
    hgt_msg<<<dim3((ET + 7) / 8), blk, 0, stream>>>(w2f, w5f, score, denom,
                                                    recv, send, agg, ET);
}

extern "C" void kernel_launch(void* const* d_in, const int* in_sizes, int n_in,
                              void* d_out, int out_size, void* d_ws, size_t ws_size,
                              hipStream_t stream) {
    const float* xa    = (const float*)d_in[0];
    const float* xo    = (const float*)d_in[1];
    const float* efeat = (const float*)d_in[2];
    const float* Wn1   = (const float*)d_in[3];
    const float* bn1   = (const float*)d_in[4];
    const float* We1   = (const float*)d_in[5];
    const float* be1   = (const float*)d_in[6];
    const float* Wn2   = (const float*)d_in[7];
    const float* bn2   = (const float*)d_in[8];
    const float* We2   = (const float*)d_in[9];
    const float* be2   = (const float*)d_in[10];
    const int*   recv  = (const int*)d_in[11];
    const int*   send  = (const int*)d_in[12];
    float* out = (float*)d_out;

    char* p = (char*)d_ws;
    float* Z    = (float*)p;  p += (size_t)NN * HD * sizeof(float);     // 32 MB
    float* w2f  = (float*)p;  p += (size_t)NN * HD * sizeof(float);     // 32 MB
    float* w3f  = (float*)p;  p += (size_t)NN * HD * sizeof(float);     // 32 MB
    float* w4f  = (float*)p;  p += (size_t)NN * HD * sizeof(float);     // 32 MB
    unsigned short* w5f = (unsigned short*)p; p += (size_t)ET * HD * 2; // 256 MB
    unsigned short* Xbf = (unsigned short*)p; p += (size_t)NN * FD * 2; // 16 MB
    unsigned short* EFbf= (unsigned short*)p; p += (size_t)ET * FD * 2; // 256 MB
    unsigned short* WnT = (unsigned short*)p; p += (size_t)8 * FD * HD * 2; // 16 MB
    unsigned short* WeT = (unsigned short*)p; p += (size_t)3 * FD * HD * 2; // 6 MB
    float* score = (float*)p; p += (size_t)ET * sizeof(float);
    float* m     = (float*)p; p += (size_t)NN * sizeof(float);
    float* denom = (float*)p; p += (size_t)NN * sizeof(float);

    const dim3 blk(256);
    // one-time bf16 conversions of GEMM A-operands
    hgt_cvt<<<dim3((unsigned)(((size_t)NA * FD / 4 + 255) / 256)), blk, 0, stream>>>(
        xa, Xbf, (long)NA * FD);
    hgt_cvt<<<dim3((unsigned)(((size_t)NA * FD / 4 + 255) / 256)), blk, 0, stream>>>(
        xo, Xbf + (size_t)NA * FD, (long)NA * FD);
    hgt_cvt<<<dim3((unsigned)(((size_t)ET * FD / 4 + 255) / 256)), blk, 0, stream>>>(
        efeat, EFbf, (long)ET * FD);

    // layer 1: inputs Xbf -> Z
    run_layer(Xbf, EFbf, Wn1, bn1, We1, be1, recv, send, WnT, WeT,
              Z, w2f, w3f, w4f, w5f, score, m, denom, stream);
    // re-convert layer-1 output for layer-2 A operand
    hgt_cvt<<<dim3((unsigned)(((size_t)NN * FD / 4 + 255) / 256)), blk, 0, stream>>>(
        Z, Xbf, (long)NN * FD);
    // layer 2: inputs Xbf -> d_out directly
    run_layer(Xbf, EFbf, Wn2, bn2, We2, be2, recv, send, WnT, WeT,
              out, w2f, w3f, w4f, w5f, score, m, denom, stream);
}
